// MambaTemporal_32873679684185
// MI455X (gfx1250) — matrix operand order repaired and checked by reading, not compiled, not run on hardware
//
#include <hip/hip_runtime.h>
#include <hip/hip_bf16.h>

#define D_MODEL  1024
#define D_STATE  16
#define D_CONV   4
#define D_INNER  2048
#define DT_RANK  64
#define B_SZ     8
#define T_LEN    512
#define NTOK     (B_SZ * T_LEN)          // 4096 tokens
#define XDBL_W   (DT_RANK + 2 * D_STATE) // 96

typedef __attribute__((ext_vector_type(16))) _Float16 v16h;
typedef __attribute__((ext_vector_type(8)))  _Float16 v8h;
typedef __attribute__((ext_vector_type(8)))  float    v8f;

// ---------------------------------------------------------------- convert
__global__ __launch_bounds__(256)
void k_f32_to_f16(const float* __restrict__ src, _Float16* __restrict__ dst, int n) {
    int i = blockIdx.x * 256 + threadIdx.x;
    if (i < n) dst[i] = (_Float16)src[i];
}

// ---------------------------------------------------------------- layernorm -> f16
__global__ __launch_bounds__(256)
void k_layernorm_f16(const float* __restrict__ x, const float* __restrict__ w,
                     const float* __restrict__ b, _Float16* __restrict__ out) {
    const int row = blockIdx.x;                       // token index
    const float* xr = x + (size_t)row * D_MODEL;
    float v[4], sum = 0.f, sq = 0.f;
#pragma unroll
    for (int j = 0; j < 4; ++j) {
        v[j] = xr[threadIdx.x + j * 256];
        sum += v[j];
        sq  += v[j] * v[j];
    }
    __shared__ float ssum[256], ssq[256];
    ssum[threadIdx.x] = sum; ssq[threadIdx.x] = sq;
    __syncthreads();
    for (int s = 128; s > 0; s >>= 1) {
        if (threadIdx.x < s) {
            ssum[threadIdx.x] += ssum[threadIdx.x + s];
            ssq [threadIdx.x] += ssq [threadIdx.x + s];
        }
        __syncthreads();
    }
    const float mu   = ssum[0] * (1.f / D_MODEL);
    const float var  = ssq[0] * (1.f / D_MODEL) - mu * mu;
    const float rstd = rsqrtf(var + 1e-5f);
    _Float16* outr = out + (size_t)row * D_MODEL;
#pragma unroll
    for (int j = 0; j < 4; ++j) {
        int c = threadIdx.x + j * 256;
        outr[c] = (_Float16)((v[j] - mu) * rstd * w[c] + b[c]);
    }
}

// ---------------------------------------------------------------- WMMA GEMM
// C[M,N] = A[M,K] * B[N,K]^T   (A row-major lda, B row-major ldb = weight (out,in))
// epilog: 1 = store f16 to Cb
//         2 = store f32 to Cf and f16 to Cb
//         3 = softplus(acc + bias[n]) -> f16 Cb
//         4 = Cf = resid + acc (f32)
//
// Block = 256 threads = 8 waves; block tile 64(M) x 128(N); each wave owns a
// 16(M) x 64(N) strip -> 4 v_wmma_f32_16x16x32_f16 per 32-deep K step.
// All tile predicates are derived via readfirstlane so they are SALU-uniform:
// WMMA always executes with EXEC all-1s (ISA requirement) and no accumulator
// phi-copies are generated.
__global__ __launch_bounds__(256)
void k_gemm_f16_wmma(const _Float16* __restrict__ A, int lda,
                     const _Float16* __restrict__ B, int ldb,
                     int M, int N, int K,
                     float* __restrict__ Cf, _Float16* __restrict__ Cb, int ldc,
                     const float* __restrict__ bias,
                     const float* __restrict__ resid,
                     int epilog)
{
    const int tid  = threadIdx.x;
    const int lane = tid & 31;
    // scalar (wave-uniform) wave index: lane 0's tid is wave*32
    const int wave = __builtin_amdgcn_readfirstlane(tid) >> 5;
    const int wm = wave & 3;           // 4 waves along M
    const int wn = wave >> 2;          // 2 waves along N
    const int m0  = blockIdx.x * 64 + wm * 16;
    const int n0b = blockIdx.y * 128 + wn * 64;
    const int mr = lane & 15;          // A row within tile / C column lane
    const int kh = lane >> 4;          // K-half select

    v8f acc[4] = {v8f{}, v8f{}, v8f{}, v8f{}};
    const _Float16* arow = A + (size_t)(m0 + mr) * lda;
    const _Float16* brow[4];
#pragma unroll
    for (int j = 0; j < 4; ++j)
        brow[j] = B + (size_t)(n0b + j * 16 + mr) * ldb;  // deref only if subtile valid

    const bool full = (n0b + 64) <= N;      // scalar condition

    if (full) {
        // hot path: branch-free, 4 WMMAs per K step
#pragma unroll 2
        for (int k0 = 0; k0 < K; k0 += 32) {
            v8h lo = *(const v8h*)(arow + k0 + kh * 16);
            v8h hi = *(const v8h*)(arow + k0 + kh * 16 + 8);
            v16h af = __builtin_shufflevector(lo, hi,
                        0,1,2,3,4,5,6,7,8,9,10,11,12,13,14,15);
#pragma unroll
            for (int j = 0; j < 4; ++j) {
                v16h bf = *(const v16h*)(brow[j] + k0 + kh * 16);
                acc[j] = __builtin_amdgcn_wmma_f32_16x16x32_f16(
                    false, af, false, bf, (short)0, acc[j], false, false);
            }
        }
    } else {
        for (int k0 = 0; k0 < K; k0 += 32) {
            v8h lo = *(const v8h*)(arow + k0 + kh * 16);
            v8h hi = *(const v8h*)(arow + k0 + kh * 16 + 8);
            v16h af = __builtin_shufflevector(lo, hi,
                        0,1,2,3,4,5,6,7,8,9,10,11,12,13,14,15);
#pragma unroll
            for (int j = 0; j < 4; ++j) {
                if (n0b + j * 16 + 16 <= N) {     // scalar guard -> s_cbranch
                    v16h bf = *(const v16h*)(brow[j] + k0 + kh * 16);
                    acc[j] = __builtin_amdgcn_wmma_f32_16x16x32_f16(
                        false, af, false, bf, (short)0, acc[j], false, false);
                }
            }
        }
    }

    // C/D layout: lane nc = lane&15 -> column, VGPR r -> row m0 + r + 8*kh
    const int nc = lane & 15;
#pragma unroll
    for (int j = 0; j < 4; ++j) {
        if (n0b + j * 16 + 16 > N) continue;      // scalar guard
        const int n = n0b + j * 16 + nc;
#pragma unroll
        for (int r = 0; r < 8; ++r) {
            const int m = m0 + r + kh * 8;
            const float v = acc[j][r];
            const size_t o = (size_t)m * ldc + n;
            if (epilog == 1) {
                Cb[o] = (_Float16)v;
            } else if (epilog == 2) {
                Cf[o] = v;
                Cb[o] = (_Float16)v;
            } else if (epilog == 3) {
                const float t  = v + bias[n];
                const float sp = (t > 20.f) ? t : log1pf(__expf(t));
                Cb[o] = (_Float16)sp;
            } else {
                Cf[o] = resid[o] + v;
            }
        }
    }
}

// ---------------------------------------------------------------- causal depthwise conv (K=4) + SiLU
__global__ __launch_bounds__(256)
void k_conv_silu(const _Float16* __restrict__ xz, const float* __restrict__ cw,
                 const float* __restrict__ cb, _Float16* __restrict__ xi)
{
    const int idx = blockIdx.x * 256 + threadIdx.x;   // over NTOK*D_INNER
    if (idx >= NTOK * D_INNER) return;
    const int d  = idx & (D_INNER - 1);
    const int bt = idx >> 11;                         // D_INNER = 2^11
    const int t  = bt & (T_LEN - 1);
    float acc = cb[d];
#pragma unroll
    for (int j = 0; j < D_CONV; ++j) {
        const int tt = t - (D_CONV - 1) + j;
        if (tt >= 0)
            acc += (float)xz[(size_t)(bt - (D_CONV - 1) + j) * (2 * D_INNER) + d]
                   * cw[d * D_CONV + j];
    }
    const float s = acc / (1.f + __expf(-acc));       // SiLU
    xi[idx] = (_Float16)s;
}

// ---------------------------------------------------------------- selective scan (sequential in T)
// one thread per (b, d) channel; 16 states in registers; fused D-skip + SiLU(res) gate
__global__ __launch_bounds__(256)
void k_scan(const _Float16* __restrict__ u,   const _Float16* __restrict__ dt,
            const float* __restrict__ xdbl,   const float* __restrict__ A_log,
            const float* __restrict__ Dsk,    const _Float16* __restrict__ xz,
            _Float16* __restrict__ yg)
{
    const int idx = blockIdx.x * 256 + threadIdx.x;   // over B_SZ*D_INNER
    if (idx >= B_SZ * D_INNER) return;
    const int d = idx & (D_INNER - 1);
    const int b = idx >> 11;

    float Ar[D_STATE], h[D_STATE];
#pragma unroll
    for (int n = 0; n < D_STATE; ++n) {
        Ar[n] = -__expf(A_log[d * D_STATE + n]);
        h[n]  = 0.f;
    }
    const float Dd = Dsk[d];

    for (int t = 0; t < T_LEN; ++t) {
        const size_t bt = (size_t)b * T_LEN + t;
        const float dtv = (float)dt[bt * D_INNER + d];
        const float uv  = (float)u [bt * D_INNER + d];
        const float* Bp = xdbl + bt * XDBL_W + DT_RANK;
        const float* Cp = Bp + D_STATE;
        const float du = dtv * uv;
        float y = 0.f;
#pragma unroll
        for (int n = 0; n < D_STATE; ++n) {
            const float dA = __expf(dtv * Ar[n]);
            h[n] = h[n] * dA + du * Bp[n];
            y += h[n] * Cp[n];
        }
        y += uv * Dd;
        const float r = (float)xz[bt * (2 * D_INNER) + D_INNER + d];
        const float g = r / (1.f + __expf(-r));
        yg[bt * D_INNER + d] = (_Float16)(y * g);
    }
}

// ---------------------------------------------------------------- launcher
extern "C" void kernel_launch(void* const* d_in, const int* in_sizes, int n_in,
                              void* d_out, int out_size, void* d_ws, size_t ws_size,
                              hipStream_t stream) {
    const float* x_in   = (const float*)d_in[0];
    const float* ln_w   = (const float*)d_in[1];
    const float* ln_b   = (const float*)d_in[2];
    const float* in_w   = (const float*)d_in[3];
    const float* conv_w = (const float*)d_in[4];
    const float* conv_b = (const float*)d_in[5];
    const float* xp_w   = (const float*)d_in[6];
    const float* dtp_w  = (const float*)d_in[7];
    const float* dtp_b  = (const float*)d_in[8];
    const float* A_log  = (const float*)d_in[9];
    const float* D_skip = (const float*)d_in[10];
    const float* out_w  = (const float*)d_in[11];
    float* outp = (float*)d_out;

    char* ws = (char*)d_ws;
    size_t off = 0;
    auto alloc = [&](size_t bytes) -> char* {
        char* p = ws + off;
        off += (bytes + 255) & ~(size_t)255;
        return p;
    };
    _Float16* xn    = (_Float16*)alloc((size_t)NTOK * D_MODEL * 2);
    _Float16* xzh   = (_Float16*)alloc((size_t)NTOK * 2 * D_INNER * 2);
    _Float16* xih   = (_Float16*)alloc((size_t)NTOK * D_INNER * 2);
    float*    xdbl  = (float*)   alloc((size_t)NTOK * XDBL_W * 4);
    _Float16* xdblh = (_Float16*)alloc((size_t)NTOK * XDBL_W * 2);
    _Float16* dth   = (_Float16*)alloc((size_t)NTOK * D_INNER * 2);
    _Float16* ygh   = (_Float16*)alloc((size_t)NTOK * D_INNER * 2);
    float*    curx  = (float*)   alloc((size_t)NTOK * D_MODEL * 4);
    _Float16* w_in  = (_Float16*)alloc((size_t)2 * D_INNER * D_MODEL * 2);
    _Float16* w_xp  = (_Float16*)alloc((size_t)XDBL_W * D_INNER * 2);
    _Float16* w_dt  = (_Float16*)alloc((size_t)D_INNER * DT_RANK * 2);
    _Float16* w_out = (_Float16*)alloc((size_t)D_MODEL * D_INNER * 2);

    for (int L = 0; L < 2; ++L) {
        const float* src = (L == 0) ? x_in : curx;   // residual / LN input
        float*       dst = (L == 1) ? outp : curx;   // residual output

        // ---- weights -> f16 (per layer slice)
        {
            const int n1 = 2 * D_INNER * D_MODEL;
            k_f32_to_f16<<<(n1 + 255) / 256, 256, 0, stream>>>(in_w + (size_t)L * n1, w_in, n1);
            const int n2 = XDBL_W * D_INNER;
            k_f32_to_f16<<<(n2 + 255) / 256, 256, 0, stream>>>(xp_w + (size_t)L * n2, w_xp, n2);
            const int n3 = D_INNER * DT_RANK;
            k_f32_to_f16<<<(n3 + 255) / 256, 256, 0, stream>>>(dtp_w + (size_t)L * n3, w_dt, n3);
            const int n4 = D_MODEL * D_INNER;
            k_f32_to_f16<<<(n4 + 255) / 256, 256, 0, stream>>>(out_w + (size_t)L * n4, w_out, n4);
        }

        // ---- layernorm
        k_layernorm_f16<<<NTOK, 256, 0, stream>>>(src, ln_w + (size_t)L * D_MODEL,
                                                  ln_b + (size_t)L * D_MODEL, xn);

        // ---- in_proj: (4096x1024)x(1024x4096) -> xz (f16)
        dim3 g1(NTOK / 64, (2 * D_INNER + 127) / 128);
        k_gemm_f16_wmma<<<g1, 256, 0, stream>>>(xn, D_MODEL, w_in, D_MODEL,
                                                NTOK, 2 * D_INNER, D_MODEL,
                                                nullptr, xzh, 2 * D_INNER,
                                                nullptr, nullptr, 1);

        // ---- causal depthwise conv + SiLU -> xi (f16)
        const int nconv = NTOK * D_INNER;
        k_conv_silu<<<(nconv + 255) / 256, 256, 0, stream>>>(
            xzh, conv_w + (size_t)L * D_INNER * D_CONV,
            conv_b + (size_t)L * D_INNER, xih);

        // ---- x_proj: (4096x2048)x(2048x96) -> x_dbl (f32 for scan B/C, f16 for dt_proj A)
        dim3 g2(NTOK / 64, 1);
        k_gemm_f16_wmma<<<g2, 256, 0, stream>>>(xih, D_INNER, w_xp, D_INNER,
                                                NTOK, XDBL_W, D_INNER,
                                                xdbl, xdblh, XDBL_W,
                                                nullptr, nullptr, 2);

        // ---- dt_proj + bias + softplus: (4096x64)x(64x2048) -> dt (f16)
        dim3 g3(NTOK / 64, (D_INNER + 127) / 128);
        k_gemm_f16_wmma<<<g3, 256, 0, stream>>>(xdblh, XDBL_W, w_dt, DT_RANK,
                                                NTOK, D_INNER, DT_RANK,
                                                nullptr, dth, D_INNER,
                                                dtp_b + (size_t)L * D_INNER, nullptr, 3);

        // ---- selective scan + D-skip + SiLU(res) gate -> yg (f16)
        const int nscan = B_SZ * D_INNER;
        k_scan<<<(nscan + 255) / 256, 256, 0, stream>>>(
            xih, dth, xdbl,
            A_log + (size_t)L * D_INNER * D_STATE,
            D_skip + (size_t)L * D_INNER, xzh, ygh);

        // ---- out_proj + residual: (4096x2048)x(2048x1024) + src -> dst (f32)
        dim3 g4(NTOK / 64, (D_MODEL + 127) / 128);
        k_gemm_f16_wmma<<<g4, 256, 0, stream>>>(ygh, D_INNER, w_out, D_INNER,
                                                NTOK, D_MODEL, D_INNER,
                                                dst, nullptr, D_MODEL,
                                                nullptr, src, 4);
    }
    (void)in_sizes; (void)n_in; (void)out_size; (void)ws_size;
}
